// LinearCrossEntropyLoss_32478542692730
// MI455X (gfx1250) — compile-verified, matrix operand order checked by reading, not run on hardware
//
#include <hip/hip_runtime.h>
#include <math.h>

// ---------------- CDNA5 / gfx1250 fused linear-cross-entropy ----------------
// Phase 0: split f32 W / acts into bf16 hi/lo planes in d_ws (once per launch,
//          ~0.5GB streamed ~= 25us) so the hot loop has ZERO conversion VALU.
// Phase 1: blocks = (128 token-blocks x 40 vocab slices). 32 tokens' hi/lo
//          planes async-copied to LDS as 16 "row pairs" so ALL 8 fragment
//          loads per k-step fold into the 16-bit DS immediate field (zero
//          in-loop LDS address VALU). 8 waves = 8 column groups, each wave
//          computes BOTH 16-token tiles per B fragment (6 WMMAs / k-step).
//          Online logsumexp in registers, 16-lane butterfly, per-slice partial.
// Phase 2: 1-block reduction -> masked mean.
// NOTE: requires ws_size >= ~300 MB (W planes 262MB + act planes 33.5MB + partials).

typedef __attribute__((ext_vector_type(16))) __bf16 v16bf;
typedef __attribute__((ext_vector_type(8)))  __bf16 v8bf;
typedef __attribute__((ext_vector_type(4)))  __bf16 v4bf;
typedef __attribute__((ext_vector_type(8)))  float  v8f;
typedef __attribute__((ext_vector_type(4)))  float  f32x4;

#define IGNORE_INDEX (-100)

constexpr int D              = 2048;
constexpr int V              = 32000;
constexpr int NTOK           = 4096;              // B*S
constexpr int TOK_PER_BLOCK  = 32;
constexpr int NUM_TOK_BLOCKS = NTOK / TOK_PER_BLOCK;   // 128
constexpr int VTILES         = V / 16;            // 2000
constexpr int SLICES         = 40;
constexpr int TILES_PER_SLICE = VTILES / SLICES;  // 50
// LDS "row pair" layout, one pair per lane row (arow 0..15):
//   [ t0-hi 2048 | t0-lo 2048 | t1-hi 2048 | t1-lo 2048 | pad 8 ] bf16
// PAIR = 8200 elems = 16400 B; 16400 % 256 == 16 -> the 16 lanes of each
// half-wave hit distinct 4-bank groups (conflict-free b128), and every
// fragment offset (max ~16398 B) fits the 16-bit DS immediate field.
constexpr int PAIR_ELEMS     = 4 * D + 8;         // 8200

// workspace byte offsets
constexpr size_t WS_WHI  = 0;
constexpr size_t WS_WLO  = WS_WHI + (size_t)V * D * 2;
constexpr size_t WS_AHI  = WS_WLO + (size_t)V * D * 2;
constexpr size_t WS_ALO  = WS_AHI + (size_t)NTOK * D * 2;
constexpr size_t WS_PART = WS_ALO + (size_t)NTOK * D * 2;

union F32x4 { f32x4 v; float e[4]; };
union F32x8 { v8f   v; float e[8]; };

// ---------------- phase 0: f32 -> bf16 hi/lo plane split ----------------
__global__ __launch_bounds__(256)
void split_bf16(const float* __restrict__ src,
                __bf16* __restrict__ hi, __bf16* __restrict__ lo, int n4)
{
    const int stride = gridDim.x * 256;
    for (int i = blockIdx.x * 256 + threadIdx.x; i < n4; i += stride) {
        F32x4 x; x.v = *(const f32x4*)(src + (size_t)i * 4);
        v4bf h, l;
        #pragma unroll
        for (int j = 0; j < 4; j++) {
            __bf16 hb = (__bf16)x.e[j];
            h[j] = hb;
            l[j] = (__bf16)(x.e[j] - (float)hb);
        }
        *(v4bf*)(hi + (size_t)i * 4) = h;
        *(v4bf*)(lo + (size_t)i * 4) = l;
    }
}

// ---------------- phase 1: fused matmul + online logsumexp ----------------
__global__ __launch_bounds__(256, 1)
void lce_partial(const __bf16* __restrict__ Whi, const __bf16* __restrict__ Wlo,
                 const __bf16* __restrict__ Ahi_g, const __bf16* __restrict__ Alo_g,
                 const int*    __restrict__ targets,
                 float*        __restrict__ partials)   // [SLICES][NTOK][3]
{
    extern __shared__ __attribute__((aligned(16))) __bf16 smem[];  // [16][PAIR_ELEMS]

    __shared__ float comb_m[2][8][16];
    __shared__ float comb_s[2][8][16];
    __shared__ float comb_t[2][8][16];

    const int tid   = threadIdx.x;
    const int lane  = tid & 31;          // wave32
    const int wave  = tid >> 5;          // column group 0..7
    const int tok0  = blockIdx.x * TOK_PER_BLOCK;
    const int slice = blockIdx.y;

    // ---- async-copy the 32-token bf16 hi/lo planes into LDS (row-pair layout) ----
    {
        const uint32_t dynbase = (__builtin_amdgcn_groupstaticsize() + 15u) & ~15u;
        const uint64_t gH = (uint64_t)(uintptr_t)(Ahi_g + (size_t)tok0 * D);
        const uint64_t gL = (uint64_t)(uintptr_t)(Alo_g + (size_t)tok0 * D);
        for (int c = tid; c < TOK_PER_BLOCK * (D / 8); c += 256) {   // b128 chunks
            int row  = c / (D / 8);          // global token row 0..31
            int c8   = c % (D / 8);
            int r    = row & 15;             // lane row (pair index)
            int tile = row >> 4;             // token tile 0/1
            uint32_t ldsH = dynbase + (uint32_t)(r * PAIR_ELEMS + tile * 2 * D + c8 * 8) * 2u;
            uint32_t ldsL = ldsH + (uint32_t)D * 2u;                 // lo half of the tile
            int voff = (row * D + c8 * 8) * 2;                       // bytes from plane base
            asm volatile("global_load_async_to_lds_b128 %0, %1, %2"
                         :: "v"(ldsH), "v"(voff), "s"(gH) : "memory");
            asm volatile("global_load_async_to_lds_b128 %0, %1, %2"
                         :: "v"(ldsL), "v"(voff), "s"(gL) : "memory");
        }
        asm volatile("s_wait_asynccnt 0x0" ::: "memory");
    }
    __syncthreads();

    // fragment addressing (ISA 7.12.2): lane&15 = row/col, lane>>4 = K-half
    const int arow = lane & 15;
    const int kh   = lane >> 4;
    const int ncol = lane & 15;
    const __bf16* rb = smem + arow * PAIR_ELEMS;   // single per-lane base for all 8 loads

    int tgi[2][8];
    #pragma unroll
    for (int r = 0; r < 8; r++) {
        tgi[0][r] = targets[tok0 + r + 8 * kh];           // C row = r + 8*(lane>>4)
        tgi[1][r] = targets[tok0 + 16 + r + 8 * kh];
    }

    float m[2][8], s[2][8], tg[2][8];
    #pragma unroll
    for (int t = 0; t < 2; t++)
        #pragma unroll
        for (int r = 0; r < 8; r++) { m[t][r] = -INFINITY; s[t][r] = 0.f; tg[t][r] = 0.f; }

    for (int ti = wave; ti < TILES_PER_SLICE; ti += 8) {
        const int v0 = (slice * TILES_PER_SLICE + ti) * 16;
        const __bf16* wrh = Whi + (size_t)(v0 + ncol) * D;
        const __bf16* wrl = Wlo + (size_t)(v0 + ncol) * D;

        v8f c0 = {0.f,0.f,0.f,0.f,0.f,0.f,0.f,0.f};
        v8f c1 = {0.f,0.f,0.f,0.f,0.f,0.f,0.f,0.f};

        for (int k0 = 0; k0 < D; k0 += 32) {
            const int kb = k0 + kh * 8;

            // B fragment: pre-split planes, pure b128 loads, no conversion
            v8bf b0h = *(const v8bf*)(wrh + kb);
            v8bf b1h = *(const v8bf*)(wrh + kb + 16);
            v8bf b0l = *(const v8bf*)(wrl + kb);
            v8bf b1l = *(const v8bf*)(wrl + kb + 16);
            v16bf Bh = __builtin_shufflevector(b0h, b1h, 0,1,2,3,4,5,6,7,8,9,10,11,12,13,14,15);
            v16bf Bl = __builtin_shufflevector(b0l, b1l, 0,1,2,3,4,5,6,7,8,9,10,11,12,13,14,15);

            // A fragments for both token tiles; all 8 loads fold into DS imm field
            v8bf x0, x1;
            x0 = *(const v8bf*)(rb + kb);             x1 = *(const v8bf*)(rb + kb + 16);
            v16bf A0h = __builtin_shufflevector(x0, x1, 0,1,2,3,4,5,6,7,8,9,10,11,12,13,14,15);
            x0 = *(const v8bf*)(rb + D + kb);         x1 = *(const v8bf*)(rb + D + kb + 16);
            v16bf A0l = __builtin_shufflevector(x0, x1, 0,1,2,3,4,5,6,7,8,9,10,11,12,13,14,15);
            x0 = *(const v8bf*)(rb + 2 * D + kb);     x1 = *(const v8bf*)(rb + 2 * D + kb + 16);
            v16bf A1h = __builtin_shufflevector(x0, x1, 0,1,2,3,4,5,6,7,8,9,10,11,12,13,14,15);
            x0 = *(const v8bf*)(rb + 3 * D + kb);     x1 = *(const v8bf*)(rb + 3 * D + kb + 16);
            v16bf A1l = __builtin_shufflevector(x0, x1, 0,1,2,3,4,5,6,7,8,9,10,11,12,13,14,15);

            // fp32-class product: Ah*Bh + Ah*Bl + Al*Bh, for both token tiles
            c0 = __builtin_amdgcn_wmma_f32_16x16x32_bf16(false, A0h, false, Bh, (short)0, c0, false, false);
            c0 = __builtin_amdgcn_wmma_f32_16x16x32_bf16(false, A0h, false, Bl, (short)0, c0, false, false);
            c0 = __builtin_amdgcn_wmma_f32_16x16x32_bf16(false, A0l, false, Bh, (short)0, c0, false, false);
            c1 = __builtin_amdgcn_wmma_f32_16x16x32_bf16(false, A1h, false, Bh, (short)0, c1, false, false);
            c1 = __builtin_amdgcn_wmma_f32_16x16x32_bf16(false, A1h, false, Bl, (short)0, c1, false, false);
            c1 = __builtin_amdgcn_wmma_f32_16x16x32_bf16(false, A1l, false, Bh, (short)0, c1, false, false);
        }

        // online softmax update with this 16-column logit tile
        F32x8 cc0, cc1; cc0.v = c0; cc1.v = c1;
        #pragma unroll
        for (int r = 0; r < 8; r++) {
            float x  = cc0.e[r];
            float nm = fmaxf(m[0][r], x);
            s[0][r] = s[0][r] * __expf(m[0][r] - nm) + __expf(x - nm);
            m[0][r] = nm;
            if (v0 + ncol == tgi[0][r]) tg[0][r] = x;

            float y   = cc1.e[r];
            float nm1 = fmaxf(m[1][r], y);
            s[1][r] = s[1][r] * __expf(m[1][r] - nm1) + __expf(y - nm1);
            m[1][r] = nm1;
            if (v0 + ncol == tgi[1][r]) tg[1][r] = y;
        }
    }

    // butterfly-reduce the 16 column lanes per row
    #pragma unroll
    for (int t = 0; t < 2; t++)
        #pragma unroll
        for (int r = 0; r < 8; r++)
            #pragma unroll
            for (int off = 1; off < 16; off <<= 1) {
                float mo = __shfl_xor(m[t][r],  off, 32);
                float so = __shfl_xor(s[t][r],  off, 32);
                float to = __shfl_xor(tg[t][r], off, 32);
                float nm = fmaxf(m[t][r], mo);
                s[t][r] = s[t][r] * __expf(m[t][r] - nm) + so * __expf(mo - nm);
                m[t][r] = nm;
                tg[t][r] += to;
            }

    if ((lane & 15) == 0) {
        #pragma unroll
        for (int t = 0; t < 2; t++)
            #pragma unroll
            for (int r = 0; r < 8; r++) {
                int row = r + 8 * kh;
                comb_m[t][wave][row] = m[t][r];
                comb_s[t][wave][row] = s[t][r];
                comb_t[t][wave][row] = tg[t][r];
            }
    }
    __syncthreads();

    // merge the 8 column-groups; one partial per (slice, token)
    if (tid < TOK_PER_BLOCK) {
        int tt2 = tid >> 4, row = tid & 15;
        float gm = -INFINITY, gs = 0.f, gt = 0.f;
        #pragma unroll
        for (int g = 0; g < 8; g++) {
            float mm = comb_m[tt2][g][row];
            float ss = comb_s[tt2][g][row];
            float nm = fmaxf(gm, mm);
            gs = gs * __expf(gm - nm) + ss * __expf(mm - nm);
            gm = nm;
            gt += comb_t[tt2][g][row];
        }
        int token = tok0 + tt2 * 16 + row;
        size_t idx = ((size_t)slice * NTOK + token) * 3;
        partials[idx + 0] = gm;
        partials[idx + 1] = gs;
        partials[idx + 2] = gt;
    }
}

// ---------------- phase 2: finalize masked mean ----------------
__global__ __launch_bounds__(256)
void lce_reduce(const float* __restrict__ partials,
                const int*   __restrict__ targets,
                float*       __restrict__ out)
{
    __shared__ float rsum[256];
    __shared__ float rcnt[256];
    const int tid = threadIdx.x;
    float total = 0.f, cnt = 0.f;

    for (int tok = tid; tok < NTOK; tok += 256) {
        float gm = -INFINITY, gs = 0.f, gt = 0.f;
        for (int sl = 0; sl < SLICES; sl++) {
            size_t idx = ((size_t)sl * NTOK + tok) * 3;
            float mm = partials[idx + 0];
            float ss = partials[idx + 1];
            float nm = fmaxf(gm, mm);
            gs = gs * __expf(gm - nm) + ss * __expf(mm - nm);
            gm = nm;
            gt += partials[idx + 2];
        }
        int t = targets[tok];
        if (t != IGNORE_INDEX) {
            total += (gm + __logf(gs)) - gt;
            cnt   += 1.f;
        }
    }
    rsum[tid] = total; rcnt[tid] = cnt;
    __syncthreads();
    for (int off = 128; off > 0; off >>= 1) {
        if (tid < off) { rsum[tid] += rsum[tid + off]; rcnt[tid] += rcnt[tid + off]; }
        __syncthreads();
    }
    if (tid == 0) out[0] = (rcnt[0] == 0.f) ? rsum[0] : rsum[0] / rcnt[0];
}

extern "C" void kernel_launch(void* const* d_in, const int* in_sizes, int n_in,
                              void* d_out, int out_size, void* d_ws, size_t ws_size,
                              hipStream_t stream) {
    (void)in_sizes; (void)n_in; (void)out_size; (void)ws_size;
    const float* acts    = (const float*)d_in[0];   // [2,2048,2048] f32
    const float* Wm      = (const float*)d_in[1];   // [32000,2048] f32
    const int*   targets = (const int*)d_in[2];     // [2,2048]
    float*       out     = (float*)d_out;

    char* ws = (char*)d_ws;
    __bf16* Whi  = (__bf16*)(ws + WS_WHI);
    __bf16* Wlo  = (__bf16*)(ws + WS_WLO);
    __bf16* Ahi  = (__bf16*)(ws + WS_AHI);
    __bf16* Alo  = (__bf16*)(ws + WS_ALO);
    float*  part = (float*) (ws + WS_PART);

    // phase 0: plane splits (pure streaming, ~25us)
    split_bf16<<<4096, 256, 0, stream>>>(Wm,   Whi, Wlo, V * D / 4);
    split_bf16<<<1024, 256, 0, stream>>>(acts, Ahi, Alo, NTOK * D / 4);

    // phase 1: fused GEMM + online logsumexp partials
    size_t smem = (size_t)16 * PAIR_ELEMS * sizeof(__bf16);  // 262400 B
    dim3 grid(NUM_TOK_BLOCKS, SLICES);
    lce_partial<<<grid, 256, smem, stream>>>(Whi, Wlo, Ahi, Alo, targets, part);

    // phase 2: finalize
    lce_reduce<<<1, 256, 0, stream>>>(part, targets, out);
}
